// GLGXCA_30279519436960
// MI455X (gfx1250) — matrix operand order repaired and checked by reading, not compiled, not run on hardware
//
#include <hip/hip_runtime.h>
#include <stdint.h>

// ---------------- problem constants ----------------
#define BB    16
#define CC    384
#define HH    56
#define WWD   56
#define HWP   3136              // 56*56
#define MTOK  (BB * HWP)        // 50176 tokens
#define HEADS 8
#define HD    48                // head dim (channels per head)

typedef __attribute__((ext_vector_type(16))) __bf16 v16bf;
typedef __attribute__((ext_vector_type(8)))  __bf16 v8bf;
typedef __attribute__((ext_vector_type(8)))  float  v8f;

__device__ __forceinline__ __bf16 f2bf(float f) {
  union { float f; uint32_t u; } c; c.f = f;
  uint32_t r = c.u + 0x7FFFu + ((c.u >> 16) & 1u);   // round-nearest-even
  union { uint16_t u; __bf16 b; } o; o.u = (uint16_t)(r >> 16);
  return o.b;
}
__device__ __forceinline__ float bf2f(__bf16 b) {
  union { uint16_t u; __bf16 b; } i; i.b = b;
  union { uint32_t u; float f; } o; o.u = ((uint32_t)i.u) << 16;
  return o.f;
}

// ---- CDNA5 async global->LDS (ASYNCcnt) helpers, inline asm (toolchain-portable) ----
__device__ __forceinline__ void async_ld_b128(uint32_t lds_addr, const void* gptr) {
  asm volatile("global_load_async_to_lds_b128 %0, %1, off"
               :: "v"(lds_addr), "v"((uint64_t)(uintptr_t)gptr)
               : "memory");
}
__device__ __forceinline__ void wait_async3() {
  asm volatile("s_wait_asynccnt 0x3" ::: "memory");
}
__device__ __forceinline__ void wait_async0() {
  asm volatile("s_wait_asynccnt 0x0" ::: "memory");
}

// ---------------- small prep kernels ----------------
__global__ void zero_stats_kernel(float* s, int n) {
  int i = blockIdx.x * 256 + threadIdx.x;
  if (i < n) s[i] = 0.0f;
}

__global__ void weights_to_bf16_kernel(const float* __restrict__ Wq,
                                       const float* __restrict__ Wk,
                                       const float* __restrict__ Wv,
                                       const float* __restrict__ Wp,
                                       __bf16* __restrict__ out) {
  int i = blockIdx.x * 256 + threadIdx.x;       // 4*384*384 total, exact
  int m = i / (CC * CC);
  int r = i - m * (CC * CC);
  const float* src = (m == 0) ? Wq : (m == 1) ? Wk : (m == 2) ? Wv : Wp;
  out[i] = f2bf(src[r]);
}

// x (B,C,HW) fp32  ->  xf (B*HW, C) bf16, tiled transpose through LDS
__global__ void transpose_to_bf16_kernel(const float* __restrict__ x,
                                         __bf16* __restrict__ xfb) {
  __shared__ float tile[32][33];
  int b  = blockIdx.z;
  int c0 = blockIdx.x * 32;
  int p0 = blockIdx.y * 32;
  int tx = threadIdx.x, ty = threadIdx.y;       // 32 x 8
  const float* xp = x + (size_t)b * CC * HWP;
  #pragma unroll
  for (int i = 0; i < 32; i += 8)
    tile[ty + i][tx] = xp[(size_t)(c0 + ty + i) * HWP + p0 + tx];
  __syncthreads();
  __bf16* op = xfb + (size_t)b * HWP * CC;
  #pragma unroll
  for (int i = 0; i < 32; i += 8)
    op[(size_t)(p0 + ty + i) * CC + c0 + tx] = f2bf(tile[tx][ty + i]);
}

// ---------------- big WMMA GEMM: (MTOK,384) x (NOUT,384)^T ----------------
// Double-buffered LDS tiles filled by GLOBAL_LOAD_ASYNC_TO_LDS_B128 (ASYNCcnt),
// consumed by v_wmma_f32_16x16x32_bf16. 3 async b128 per thread per stage.
// Mainloop fully unrolled (12 K-steps) so buffer parity / tail waits are
// compile-time and the scheduler can interleave ds_loads with WMMAs.
// QKV=true : fused Wq|Wk|Wvc (NOUT=1152). q,k stored transposed bf16 (B,C,N),
//            vc stored transposed fp32 (B,C,N).
// QKV=false: proj (NOUT=384), fuses +bias +residual, stores NCHW fp32.
template <bool QKV>
__global__ __launch_bounds__(256) void gemm_wmma_kernel(
    const __bf16* __restrict__ A,    // (MTOK, 384) bf16
    const __bf16* __restrict__ Wb,   // (NOUT, 384) bf16 row-major
    __bf16* __restrict__ qT, __bf16* __restrict__ kT, float* __restrict__ vcT,
    const float* __restrict__ bias, const float* __restrict__ xres,
    float* __restrict__ outp) {
  __shared__ __align__(16) __bf16 Asm[2][128 * 32];
  __shared__ __align__(16) __bf16 Bsm[2][64 * 32];

  const int tid  = threadIdx.x;
  const int wv   = tid >> 5;
  const int lane = tid & 31;
  const int m0   = blockIdx.y * 128;
  const int n0   = blockIdx.x * 64;
  const int mw   = (wv & 3) * 32;
  const int nw   = (wv >> 2) * 32;
  const int r    = lane & 15;
  const int hiA  = (lane & 16) ? 8  : 0;   // A: K elements split 0..7 / 16..23 per half-wave
  const int hiB  = (lane & 16) ? 8  : 0;   // B: K elements split 0..7 / 16..23 per half-wave

  // per-thread chunk coordinates (3 x 16B per stage)
  const int a0row = (tid      ) >> 2, a0seg = (tid      ) & 3;
  const int a1row = (tid + 256) >> 2, a1seg = (tid + 256) & 3;
  const int brow  = tid >> 2,         bseg  = tid & 3;

  auto issue_stage = [&](int buf, int k0) {
    async_ld_b128((uint32_t)(uintptr_t)&Asm[buf][a0row * 32 + a0seg * 8],
                  A + (size_t)(m0 + a0row) * CC + k0 + a0seg * 8);
    async_ld_b128((uint32_t)(uintptr_t)&Asm[buf][a1row * 32 + a1seg * 8],
                  A + (size_t)(m0 + a1row) * CC + k0 + a1seg * 8);
    async_ld_b128((uint32_t)(uintptr_t)&Bsm[buf][brow * 32 + bseg * 8],
                  Wb + (size_t)(n0 + brow) * CC + k0 + bseg * 8);
  };

  v8f acc[2][2] = {};
  issue_stage(0, 0);

  const int KSTEPS = CC / 32;      // 12
  #pragma unroll
  for (int ks = 0; ks < KSTEPS; ++ks) {
    if (ks + 1 < KSTEPS) {
      issue_stage((ks + 1) & 1, (ks + 1) * 32);
      wait_async3();               // stage ks complete (in-order), ks+1 in flight
    } else {
      wait_async0();
    }
    __syncthreads();               // all waves' fills visible

    const __bf16* Ab = Asm[ks & 1];
    const __bf16* Bb = Bsm[ks & 1];
    v16bf af[2], bfm[2];
    #pragma unroll
    for (int mi = 0; mi < 2; ++mi) {
      const __bf16* ap = &Ab[(mw + mi * 16 + r) * 32 + hiA];
      v8bf lo = *(const v8bf*)ap;
      v8bf hi = *(const v8bf*)(ap + 16);
      af[mi]  = __builtin_shufflevector(lo, hi, 0,1,2,3,4,5,6,7,8,9,10,11,12,13,14,15);
    }
    #pragma unroll
    for (int ni = 0; ni < 2; ++ni) {
      const __bf16* bp = &Bb[(nw + ni * 16 + r) * 32 + hiB];
      v8bf lo = *(const v8bf*)bp;
      v8bf hi = *(const v8bf*)(bp + 16);
      bfm[ni] = __builtin_shufflevector(lo, hi, 0,1,2,3,4,5,6,7,8,9,10,11,12,13,14,15);
    }
    #pragma unroll
    for (int mi = 0; mi < 2; ++mi)
      #pragma unroll
      for (int ni = 0; ni < 2; ++ni)
        acc[mi][ni] = __builtin_amdgcn_wmma_f32_16x16x32_bf16(
            false, af[mi], false, bfm[ni], (short)0, acc[mi][ni], false, false);
    __syncthreads();               // protect buffer before next overwrite
  }

  // epilogue: each lane holds 8 consecutive tokens (M) at one out-channel (N)
  #pragma unroll
  for (int mi = 0; mi < 2; ++mi) {
    int mrow = m0 + mw + mi * 16 + ((lane >> 4) << 3);  // 8-aligned token base
    int b    = mrow / HWP;
    int nt   = mrow - b * HWP;                          // HWP % 8 == 0 -> same b
    #pragma unroll
    for (int ni = 0; ni < 2; ++ni) {
      int ng = n0 + nw + ni * 16 + (lane & 15);
      if (QKV) {
        int mat = ng / CC;
        int cch = ng - mat * CC;
        size_t base = ((size_t)b * CC + cch) * HWP + nt;
        if (mat < 2) {
          __align__(16) __bf16 tmp[8];
          #pragma unroll
          for (int i = 0; i < 8; ++i) tmp[i] = f2bf(acc[mi][ni][i]);
          *(uint4*)((mat == 0 ? qT : kT) + base) = *(const uint4*)tmp;
        } else {
          float4 v0 = make_float4(acc[mi][ni][0], acc[mi][ni][1], acc[mi][ni][2], acc[mi][ni][3]);
          float4 v1 = make_float4(acc[mi][ni][4], acc[mi][ni][5], acc[mi][ni][6], acc[mi][ni][7]);
          *(float4*)(vcT + base)     = v0;
          *(float4*)(vcT + base + 4) = v1;
        }
      } else {
        size_t base = ((size_t)b * CC + ng) * HWP + nt;
        float bb = bias[ng];
        float4 r0 = *(const float4*)(xres + base);
        float4 r1 = *(const float4*)(xres + base + 4);
        float4 o0 = make_float4(acc[mi][ni][0] + bb + r0.x, acc[mi][ni][1] + bb + r0.y,
                                acc[mi][ni][2] + bb + r0.z, acc[mi][ni][3] + bb + r0.w);
        float4 o1 = make_float4(acc[mi][ni][4] + bb + r1.x, acc[mi][ni][5] + bb + r1.y,
                                acc[mi][ni][6] + bb + r1.z, acc[mi][ni][7] + bb + r1.w);
        *(float4*)(outp + base)     = o0;
        *(float4*)(outp + base + 4) = o1;
      }
    }
  }
}

// ---------------- depthwise conv helpers / BN ----------------
__device__ __forceinline__ float dwconv_at(const float* __restrict__ xp,
                                           const float* __restrict__ k9,
                                           int h, int w) {
  float acc = 0.0f;
  #pragma unroll
  for (int kh = 0; kh < 3; ++kh) {
    int hy = h + kh - 1;
    if (hy < 0 || hy >= HH) continue;
    #pragma unroll
    for (int kw = 0; kw < 3; ++kw) {
      int wx = w + kw - 1;
      if (wx < 0 || wx >= WWD) continue;
      acc += k9[kh * 3 + kw] * xp[hy * WWD + wx];
    }
  }
  return acc;
}

// pass 1: per-channel sum / sumsq of conv output (batch stats)
__global__ void conv_stats_kernel(const float* __restrict__ x,
                                  const float* __restrict__ kern,
                                  float* __restrict__ stats) {
  int b = blockIdx.z, c = blockIdx.y;
  int tid = threadIdx.x;
  int p = blockIdx.x * 256 + tid;
  const float* xp = x + ((size_t)b * CC + c) * HWP;
  const float* k9 = kern + c * 9;
  float g = 0.0f;
  if (p < HWP) g = dwconv_at(xp, k9, p / WWD, p % WWD);
  __shared__ float s1[256], s2[256];
  s1[tid] = g; s2[tid] = g * g;
  __syncthreads();
  for (int s = 128; s > 0; s >>= 1) {
    if (tid < s) { s1[tid] += s1[tid + s]; s2[tid] += s2[tid + s]; }
    __syncthreads();
  }
  if (tid == 0) {
    atomicAdd(&stats[c],      s1[0]);
    atomicAdd(&stats[CC + c], s2[0]);
  }
}

// pass 2: recompute conv, BN(train) + SiLU, gate vcT in place
__global__ void bn_silu_gate_kernel(const float* __restrict__ x,
                                    const float* __restrict__ kern,
                                    const float* __restrict__ stats,
                                    const float* __restrict__ gamma,
                                    const float* __restrict__ beta,
                                    float* __restrict__ vcT) {
  int b = blockIdx.z, c = blockIdx.y;
  int p = blockIdx.x * 256 + threadIdx.x;
  if (p >= HWP) return;
  const float inv_n = 1.0f / (float)(BB * HWP);
  float mean  = stats[c] * inv_n;
  float var   = stats[CC + c] * inv_n - mean * mean;
  float scale = gamma[c] * rsqrtf(var + 1e-5f);
  const float* xp = x + ((size_t)b * CC + c) * HWP;
  float g = dwconv_at(xp, kern + c * 9, p / WWD, p % WWD);
  g = (g - mean) * scale + beta[c];
  g = g * (1.0f / (1.0f + __expf(-g)));             // SiLU
  vcT[((size_t)b * CC + c) * HWP + p] *= g;
}

// ---------------- L2 norms of q,k rows over N ----------------
__global__ void l2norms_kernel(const __bf16* __restrict__ qT,
                               const __bf16* __restrict__ kT,
                               float* __restrict__ norms) {
  const __bf16* src = (blockIdx.y == 0) ? qT : kT;
  size_t base = (size_t)blockIdx.x * HWP;
  int tid = threadIdx.x;
  float s = 0.0f;
  for (int i = tid; i < HWP; i += 128) {
    float v = bf2f(src[base + i]);
    s += v * v;
  }
  __shared__ float red[128];
  red[tid] = s;
  __syncthreads();
  for (int st = 64; st > 0; st >>= 1) {
    if (tid < st) red[tid] += red[tid + st];
    __syncthreads();
  }
  if (tid == 0)
    norms[blockIdx.y * (BB * CC) + blockIdx.x] = fmaxf(sqrtf(red[0]), 1e-12f);
}

// ---------------- gram (48x48 over N=3136) + softmax, per (b,h) ----------------
__global__ __launch_bounds__(96) void gram_softmax_kernel(
    const __bf16* __restrict__ qT, const __bf16* __restrict__ kT,
    const float* __restrict__ norms, const float* __restrict__ temperature,
    float* __restrict__ attn) {
  int bh = blockIdx.x;
  int b = bh >> 3, h = bh & 7;
  int tid = threadIdx.x, wv = tid >> 5, lane = tid & 31;
  const __bf16* qb = qT + ((size_t)b * CC + h * HD) * HWP;
  const __bf16* kb = kT + ((size_t)b * CC + h * HD) * HWP;

  const int r   = lane & 15;
  const int hiA = (lane & 16) ? 8  : 0;
  const int hiB = (lane & 16) ? 8  : 0;
  v8f acc0 = {}, acc1 = {}, acc2 = {};
  const int d = 16 * wv + r;                 // A row for this lane

  for (int n0 = 0; n0 < HWP; n0 += 32) {
    const __bf16* ap = qb + (size_t)d * HWP + n0 + hiA;
    v8bf alo = *(const v8bf*)ap;
    v8bf ahi = *(const v8bf*)(ap + 16);
    v16bf af = __builtin_shufflevector(alo, ahi, 0,1,2,3,4,5,6,7,8,9,10,11,12,13,14,15);
    #pragma unroll
    for (int t = 0; t < 3; ++t) {
      const __bf16* bp = kb + (size_t)(16 * t + r) * HWP + n0 + hiB;
      v8bf blo = *(const v8bf*)bp;
      v8bf bhi = *(const v8bf*)(bp + 16);
      v16bf bfv = __builtin_shufflevector(blo, bhi, 0,1,2,3,4,5,6,7,8,9,10,11,12,13,14,15);
      if (t == 0)      acc0 = __builtin_amdgcn_wmma_f32_16x16x32_bf16(false, af, false, bfv, (short)0, acc0, false, false);
      else if (t == 1) acc1 = __builtin_amdgcn_wmma_f32_16x16x32_bf16(false, af, false, bfv, (short)0, acc1, false, false);
      else             acc2 = __builtin_amdgcn_wmma_f32_16x16x32_bf16(false, af, false, bfv, (short)0, acc2, false, false);
    }
  }

  __shared__ float S[48][48];
  int mrow = 16 * wv + ((lane >> 4) << 3);
  int ncol = lane & 15;
  #pragma unroll
  for (int i = 0; i < 8; ++i) {
    S[mrow + i][ncol]      = acc0[i];
    S[mrow + i][16 + ncol] = acc1[i];
    S[mrow + i][32 + ncol] = acc2[i];
  }
  __syncthreads();

  if (tid < HD) {
    int dd = tid;
    float nq   = norms[b * CC + h * HD + dd];
    float temp = temperature[h];
    float row[48];
    float mx = -3.4e38f;
    #pragma unroll 8
    for (int e = 0; e < HD; ++e) {
      float nk = norms[BB * CC + b * CC + h * HD + e];
      float v = S[dd][e] * temp / (nq * nk);
      row[e] = v;
      mx = fmaxf(mx, v);
    }
    float sum = 0.0f;
    #pragma unroll 8
    for (int e = 0; e < HD; ++e) { row[e] = __expf(row[e] - mx); sum += row[e]; }
    float inv = 1.0f / sum;
    float* op = attn + ((size_t)bh * HD + dd) * HD;
    #pragma unroll 8
    for (int e = 0; e < HD; ++e) op[e] = row[e] * inv;
  }
}

// ---------------- out = attn @ v  -> (B,N,C) bf16 ----------------
__global__ __launch_bounds__(256) void attn_v_kernel(
    const float* __restrict__ attn, const float* __restrict__ vg,
    __bf16* __restrict__ ovb) {
  int bh = blockIdx.y;
  int b = bh >> 3, h = bh & 7;
  __shared__ float As[48][48];
  for (int i = threadIdx.x; i < HD * HD; i += 256)
    As[i / HD][i % HD] = attn[(size_t)bh * HD * HD + i];
  __syncthreads();
  int n = blockIdx.x * 256 + threadIdx.x;
  if (n >= HWP) return;
  const float* vb = vg + ((size_t)b * CC + h * HD) * HWP + n;
  float o[48];
  #pragma unroll
  for (int d = 0; d < HD; ++d) o[d] = 0.0f;
  for (int e = 0; e < HD; ++e) {
    float vv = vb[(size_t)e * HWP];
    #pragma unroll
    for (int d = 0; d < HD; ++d) o[d] += As[d][e] * vv;   // broadcast LDS read
  }
  __bf16* op = ovb + ((size_t)b * HWP + n) * CC + h * HD;
  #pragma unroll
  for (int d = 0; d < HD; ++d) op[d] = f2bf(o[d]);
}

// ---------------- launcher ----------------
extern "C" void kernel_launch(void* const* d_in, const int* in_sizes, int n_in,
                              void* d_out, int out_size, void* d_ws, size_t ws_size,
                              hipStream_t stream) {
  const float* x     = (const float*)d_in[0];
  const float* Wq    = (const float*)d_in[1];
  const float* Wk    = (const float*)d_in[2];
  const float* Wv    = (const float*)d_in[3];
  const float* dwk   = (const float*)d_in[4];
  const float* gamma = (const float*)d_in[5];
  const float* beta  = (const float*)d_in[6];
  const float* temp  = (const float*)d_in[7];
  const float* Wp    = (const float*)d_in[8];
  const float* bp    = (const float*)d_in[9];
  float* out = (float*)d_out;
  char* ws = (char*)d_ws;

  const size_t szW   = (size_t)4 * CC * CC * sizeof(__bf16);    // fused Wq|Wk|Wvc|Wproj
  const size_t szXF  = (size_t)MTOK * CC * sizeof(__bf16);
  const size_t szQT  = (size_t)BB * CC * HWP * sizeof(__bf16);
  const size_t szVC  = (size_t)BB * CC * HWP * sizeof(float);
  const size_t szST  = (size_t)2 * CC * sizeof(float);
  const size_t szNRM = (size_t)2 * BB * CC * sizeof(float);

  size_t oW   = 0;
  size_t oXF  = oW + szW;
  size_t oQT  = oXF + szXF;
  size_t oKT  = oQT + szQT;
  size_t oVC  = oKT + szQT;
  size_t oST  = oVC + szVC;
  size_t oNRM = oST + szST;
  size_t oATT = oNRM + szNRM;

  __bf16* Wb    = (__bf16*)(ws + oW);
  __bf16* xfb   = (__bf16*)(ws + oXF);
  __bf16* ovb   = xfb;                      // alias: xf dead after QKV GEMM
  __bf16* qT    = (__bf16*)(ws + oQT);
  __bf16* kT    = (__bf16*)(ws + oKT);
  float*  vcT   = (float*)(ws + oVC);
  float*  stats = (float*)(ws + oST);
  float*  norms = (float*)(ws + oNRM);
  float*  attn  = (float*)(ws + oATT);

  // 0) zero BN stat accumulators
  zero_stats_kernel<<<3, 256, 0, stream>>>(stats, 2 * CC);
  // 1) weights -> bf16 fused buffer
  weights_to_bf16_kernel<<<(4 * CC * CC) / 256, 256, 0, stream>>>(Wq, Wk, Wv, Wp, Wb);
  // 2) x NCHW fp32 -> (B*N, C) bf16
  transpose_to_bf16_kernel<<<dim3(CC / 32, HWP / 32, BB), dim3(32, 8), 0, stream>>>(x, xfb);
  // 3) fused QKV GEMM (WMMA bf16, async double-buffered LDS), transposed stores
  gemm_wmma_kernel<true><<<dim3((3 * CC) / 64, MTOK / 128), 256, 0, stream>>>(
      xfb, Wb, qT, kT, vcT, nullptr, nullptr, nullptr);
  // 4) depthwise conv batch stats
  conv_stats_kernel<<<dim3((HWP + 255) / 256, CC, BB), 256, 0, stream>>>(x, dwk, stats);
  // 5) per-(b,c) L2 norms of q and k over N
  l2norms_kernel<<<dim3(BB * CC, 2), 128, 0, stream>>>(qT, kT, norms);
  // 6) BN + SiLU gate applied to v (in place)
  bn_silu_gate_kernel<<<dim3((HWP + 255) / 256, CC, BB), 256, 0, stream>>>(
      x, dwk, stats, gamma, beta, vcT);
  // 7) gram matrix (WMMA over N) + temperature/norm scaling + softmax
  gram_softmax_kernel<<<BB * HEADS, 96, 0, stream>>>(qT, kT, norms, temp, attn);
  // 8) apply attention to gated v -> (B,N,C) bf16
  attn_v_kernel<<<dim3((HWP + 255) / 256, BB * HEADS), 256, 0, stream>>>(attn, vcT, ovb);
  // 9) proj GEMM (WMMA bf16) + bias + residual -> NCHW fp32 output
  gemm_wmma_kernel<false><<<dim3(CC / 64, MTOK / 128), 256, 0, stream>>>(
      ovb, Wb + (size_t)3 * CC * CC, nullptr, nullptr, nullptr, bp, x, out);
}
